// EvoformerBlock_31361851195883
// MI455X (gfx1250) — matrix operand order repaired and checked by reading, not compiled, not run on hardware
//
#include <hip/hip_runtime.h>

// ---------------------------------------------------------------------------
// Evoformer block for MI455X (gfx1250, wave32, WMMA).
// All matmuls via V_WMMA_F32_16X16X32_F16 (f16 in, f32 accum). Activations at
// GEMM boundaries are stored f16 (producers emit f16); residuals, gates and
// LN statistics stay fp32. Weights are pre-transposed to f16 [N][K] once.
// ---------------------------------------------------------------------------

typedef _Float16 half_t;
typedef __attribute__((ext_vector_type(16))) _Float16 v16h;
typedef __attribute__((ext_vector_type(8)))  _Float16 v8h;
typedef __attribute__((ext_vector_type(8)))  float    v8f;

__device__ __forceinline__ v8f wmma32(v16h a, v16h b, v8f c) {
  // (neg_a, A, neg_b, B, c_mod, C, reuse_a, reuse_b)
  return __builtin_amdgcn_wmma_f32_16x16x32_f16(false, a, false, b, (short)0, c,
                                                false, false);
}

// A-fragment (16x32 f16) from a row-major half tile (ld halfs/row, 16B-aligned
// rows). ISA layout: lanes 0-15 -> M=lane, K {0..7,16..23}; lanes 16-31 ->
// M=lane-16, K {8..15,24..31}; v16h element e lives in VGPR e/2.
__device__ __forceinline__ v16h frag_a_lds(const half_t* p, int ld) {
  int lane = threadIdx.x & 31;
  int r = lane & 15, hi = lane >> 4;
  const half_t* q = p + r * ld;
  v16h f;
#pragma unroll
  for (int e = 0; e < 8; ++e) f[e] = q[hi * 8 + e];
#pragma unroll
  for (int e = 0; e < 8; ++e) f[8 + e] = q[16 + hi * 8 + e];
  return f;
}

// Same A-fragment straight from global f16 (row-major, ld halfs).
__device__ __forceinline__ v16h frag_a_g16(const half_t* p, long ld) {
  int lane = threadIdx.x & 31;
  int r = lane & 15, hi = lane >> 4;
  const half_t* q = p + (long)r * ld;
  v16h f;
#pragma unroll
  for (int e = 0; e < 8; ++e) f[e] = q[hi * 8 + e];
#pragma unroll
  for (int e = 0; e < 8; ++e) f[8 + e] = q[16 + hi * 8 + e];
  return f;
}

// B-fragment (32x16 f16) from a row-major tile of B^T (rows = N columns of B).
// ISA layout: lane holds column N=lane&15; lanes 0-15 cover K {0..7,16..23},
// lanes 16-31 cover K {8..15,24..31} (same K order as the A-fragment).
__device__ __forceinline__ v16h frag_b_lds(const half_t* p, int ld) {
  int lane = threadIdx.x & 31;
  int n = lane & 15, hi = lane >> 4;
  const half_t* q = p + n * ld;
  v16h f;
#pragma unroll
  for (int e = 0; e < 8; ++e) f[e] = q[hi * 8 + e];
#pragma unroll
  for (int e = 0; e < 8; ++e) f[8 + e] = q[16 + hi * 8 + e];
  return f;
}

// ---------------------------------------------------------------------------
// Generic WMMA GEMM, f16 operands: C[m,n] = epi(alpha * sum_k A[m,k]*B[k,n])
//  A[m,k]  = A[batch*sAb + m*sAm + k*sAk]       (opm==1: OPM reshape index)
//  B[k,n]  = Bt[batch*sBb + n*sBtn + k*sBtk]    (B given transposed)
//  outputs: Cf (fp32) and/or Ch (f16), ld = ldC, batch stride sCb
//  epilogue: *alpha, +bias[n] (fp32), act(0,1=sigmoid,2=relu), *mul[m*ldC+n]
// Block 128 threads (4 waves), 64x64 tile, K step 32, K%32==0.
// Unit-stride operands take a vectorized b128 staging path with prefetch.
// ---------------------------------------------------------------------------
__global__ __launch_bounds__(128) void k_gemm(
    const half_t* __restrict__ A, long sAm, long sAk, long sAb,
    const half_t* __restrict__ Bt, long sBtn, long sBtk, long sBb,
    float* __restrict__ Cf, half_t* __restrict__ Ch, long ldC, long sCb,
    const float* __restrict__ bias, const float* __restrict__ mul,
    int M, int N, int K, float alpha, int act, int opm)
{
  __shared__ __align__(16) half_t As[64 * 48];  // [m][k], k padded to 48
  __shared__ __align__(16) half_t Bs[64 * 48];  // B^T tile: [n][k]
  const int tid = threadIdx.x, wave = tid >> 5, lane = tid & 31;
  const int n0 = blockIdx.x * 64, m0 = blockIdx.y * 64;
  const long batch = blockIdx.z;
  const half_t* Ab = A + batch * sAb;
  const half_t* Bb = Bt + batch * sBb;
  const bool fastA = (!opm) && sAk == 1 && ((sAm & 7) == 0) && (m0 + 64 <= M);
  const bool fastB = sBtk == 1 && ((sBtn & 7) == 0) && (n0 + 64 <= N);
  v8f acc[4] = {};

  for (int k0 = 0; k0 < K; k0 += 32) {
    if (fastA) {
      int row = tid >> 1, part = tid & 1;
      const half_t* src = Ab + (long)(m0 + row) * sAm + k0 + part * 16;
      v8h x0 = *(const v8h*)src;
      v8h x1 = *(const v8h*)(src + 8);
      *(v8h*)&As[row * 48 + part * 16] = x0;
      *(v8h*)&As[row * 48 + part * 16 + 8] = x1;
      if (k0 + 32 < K) __builtin_prefetch(src + 32, 0, 0);
    } else {
      for (int idx = tid; idx < 2048; idx += 128) {
        int mm = idx >> 5, kk = idx & 31;
        int gm = m0 + mm, gk = k0 + kk;
        half_t v = (half_t)0;
        if (gm < M) {
          long ai;
          if (opm)  // row=(i,j), k=(c,d) -> O[(i*32+c)][(j*32+d)], ld 8192
            ai = ((long)(gm >> 8) * 32 + (gk >> 5)) * 8192 + (long)(gm & 255) * 32 + (gk & 31);
          else
            ai = (long)gm * sAm + (long)gk * sAk;
          v = Ab[ai];
        }
        As[mm * 48 + kk] = v;
      }
    }
    if (fastB) {
      int row = tid >> 1, part = tid & 1;
      const half_t* src = Bb + (long)(n0 + row) * sBtn + k0 + part * 16;
      v8h x0 = *(const v8h*)src;
      v8h x1 = *(const v8h*)(src + 8);
      *(v8h*)&Bs[row * 48 + part * 16] = x0;
      *(v8h*)&Bs[row * 48 + part * 16 + 8] = x1;
      if (k0 + 32 < K) __builtin_prefetch(src + 32, 0, 0);
    } else {
      for (int idx = tid; idx < 2048; idx += 128) {
        int kk = idx >> 6, nn = idx & 63;
        int gk = k0 + kk, gn = n0 + nn;
        Bs[nn * 48 + kk] = (gn < N) ? Bb[(long)gn * sBtn + (long)gk * sBtk] : (half_t)0;
      }
    }
    __syncthreads();
    v16h af = frag_a_lds(As + wave * 16 * 48, 48);
#pragma unroll
    for (int nt = 0; nt < 4; ++nt) {
      v16h bf = frag_b_lds(Bs + nt * 16 * 48, 48);
      acc[nt] = wmma32(af, bf, acc[nt]);
    }
    __syncthreads();
  }

  // C/D layout: lane -> n = lane&15, m = vgpr + 8*(lane>>4)
  const int nl = lane & 15, hi = lane >> 4;
#pragma unroll
  for (int nt = 0; nt < 4; ++nt)
#pragma unroll
    for (int v = 0; v < 8; ++v) {
      int gm = m0 + wave * 16 + hi * 8 + v;
      int gn = n0 + nt * 16 + nl;
      if (gm < M && gn < N) {
        float val = acc[nt][v] * alpha;
        if (bias) val += bias[gn];
        if (act == 1) val = 1.f / (1.f + __expf(-val));
        else if (act == 2) val = fmaxf(val, 0.f);
        if (mul) val *= mul[(long)gm * ldC + gn];
        if (Cf) Cf[batch * sCb + (long)gm * ldC + gn] = val;
        if (Ch) Ch[batch * sCb + (long)gm * ldC + gn] = (half_t)val;
      }
    }
}

// ---------------------------------------------------------------------------
// Fused gated multi-head attention (head dim 32, seq len N in {64,256}).
// grid = (N/64 q-tiles, H, batch); block = 128 (4 waves x 16 query rows).
// Q/K/V f16 (Q pre-scaled by 1/sqrt(32)); gate G fp32 (already sigmoid);
// bias fp32 [H][N][N] or null; output O f16.
// ---------------------------------------------------------------------------
__global__ __launch_bounds__(128) void k_attn(
    const half_t* __restrict__ Q, const half_t* __restrict__ Kx,
    const half_t* __restrict__ V, const float* __restrict__ G,
    const float* __restrict__ bias, half_t* __restrict__ O,
    int N, int HC)
{
  extern __shared__ char smem[];
  const int tid = threadIdx.x, lane = tid & 31, wave = tid >> 5;
  const int q0 = blockIdx.x * 64;
  const int h = blockIdx.y;
  const long b = blockIdx.z;

  half_t* Ksh = (half_t*)smem;                 // [N][48]
  half_t* Vt  = Ksh + (long)N * 48;            // [32][N+16]  (V^T)
  size_t off = ((size_t)N * 48 + (size_t)32 * (N + 16)) * sizeof(half_t);
  off = (off + 15) & ~(size_t)15;
  float* Sw = (float*)(smem + off);            // 4 waves x [16][N]
  const int Vs = N + 16;

  // K rows: vectorized contiguous copy; V: scalar transpose into Vt.
  for (int idx = tid; idx < N * 2; idx += 128) {
    int row = idx >> 1, part = idx & 1;
    const half_t* src = Kx + (b * N + row) * (long)HC + (long)h * 32 + part * 16;
    v8h x0 = *(const v8h*)src;
    v8h x1 = *(const v8h*)(src + 8);
    *(v8h*)&Ksh[row * 48 + part * 16] = x0;
    *(v8h*)&Ksh[row * 48 + part * 16 + 8] = x1;
  }
  for (int idx = tid; idx < N * 32; idx += 128) {
    int k = idx >> 5, c = idx & 31;
    Vt[c * Vs + k] = V[(b * N + k) * (long)HC + (long)h * 32 + c];
  }
  __syncthreads();

  const half_t* qb = Q + (b * N + q0 + wave * 16) * (long)HC + (long)h * 32;
  v16h qf = frag_a_g16(qb, HC);

  float* S = Sw + wave * 16 * N;
  const int ntiles = N >> 4;
  for (int nt = 0; nt < ntiles; ++nt) {
    v16h bf = frag_b_lds(Ksh + nt * 16 * 48, 48);
    v8f acc = {};
    acc = wmma32(qf, bf, acc);
    const int nl = lane & 15, hi = lane >> 4;
#pragma unroll
    for (int v = 0; v < 8; ++v)
      S[(hi * 8 + v) * N + nt * 16 + nl] = acc[v];
  }

  // Wave-synchronous softmax per row; rewrite P as f16 in place over S.
  half_t* P = (half_t*)S;
  const int Ps = N + 16;
  const int nper = N >> 5;   // <= 8
  for (int r = 0; r < 16; ++r) {
    float vv[8];
    float mx = -3.0e38f;
    for (int i = 0; i < nper; ++i) {
      int j = i * 32 + lane;
      float x = S[r * N + j];
      if (bias) x += bias[((long)h * N + q0 + wave * 16 + r) * (long)N + j];
      vv[i] = x; mx = fmaxf(mx, x);
    }
    for (int o = 16; o; o >>= 1) mx = fmaxf(mx, __shfl_xor(mx, o, 32));
    float sum = 0.f;
    for (int i = 0; i < nper; ++i) { vv[i] = __expf(vv[i] - mx); sum += vv[i]; }
    for (int o = 16; o; o >>= 1) sum += __shfl_xor(sum, o, 32);
    float inv = 1.f / sum;
    for (int i = 0; i < nper; ++i) P[r * Ps + i * 32 + lane] = (half_t)(vv[i] * inv);
  }

  // O = P (16xN) * V (Nx32)
  v8f oacc[2] = {};
  for (int ks = 0; ks < (N >> 5); ++ks) {
    v16h pf = frag_a_lds(P + ks * 32, Ps);
#pragma unroll
    for (int nt = 0; nt < 2; ++nt) {
      v16h vf = frag_b_lds(Vt + nt * 16 * Vs + ks * 32, Vs);
      oacc[nt] = wmma32(pf, vf, oacc[nt]);
    }
  }
  const int nl = lane & 15, hi = lane >> 4;
#pragma unroll
  for (int nt = 0; nt < 2; ++nt)
#pragma unroll
    for (int v = 0; v < 8; ++v) {
      long row = b * N + q0 + wave * 16 + hi * 8 + v;
      long g = row * HC + (long)h * 32 + nt * 16 + nl;
      O[g] = (half_t)(G[g] * oacc[nt][v]);   // G already sigmoid(gate)
    }
}

// ---------------------------------------------------------------------------
// LayerNorm over last dim C (biased var, eps 1e-5), fp32 in -> f16 out.
// Input element (row, c) at x[(row/Nn)*sB + (row%Nn)*sN + c*sC].
// ---------------------------------------------------------------------------
__global__ __launch_bounds__(256) void k_ln(
    const float* __restrict__ x, const float* __restrict__ g,
    const float* __restrict__ bv, half_t* __restrict__ out,
    int rows, int Nn, long sB, long sN, long sC, int C)
{
  int row = blockIdx.x * 8 + (threadIdx.x >> 5);
  if (row >= rows) return;
  int lane = threadIdx.x & 31;
  long bb = row / Nn, nn = row % Nn;
  const float* xp = x + bb * sB + nn * sN;
  float s = 0.f, s2 = 0.f;
  for (int c = lane; c < C; c += 32) {
    float v = xp[(long)c * sC]; s += v; s2 += v * v;
  }
  for (int o = 16; o; o >>= 1) { s += __shfl_xor(s, o, 32); s2 += __shfl_xor(s2, o, 32); }
  float mu = s / C, var = s2 / C - mu * mu, rs = rsqrtf(var + 1e-5f);
  for (int c = lane; c < C; c += 32) {
    float v = xp[(long)c * sC];
    out[(long)row * C + c] = (half_t)((v - mu) * rs * g[c] + bv[c]);
  }
}

// bias[h][i][j] = sum_c zn[(i*R+j)*C + c] * wb[c*H + h]   (zn f16)
__global__ void k_bias(const half_t* __restrict__ zn, const float* __restrict__ wb,
                       float* __restrict__ bias, int H, int R, int C)
{
  long idx = (long)blockIdx.x * 256 + threadIdx.x;
  long tot = (long)H * R * R;
  if (idx >= tot) return;
  int h = (int)(idx / ((long)R * R));
  long ij = idx % ((long)R * R);
  float s = 0.f;
  for (int c = 0; c < C; ++c) s += (float)zn[ij * C + c] * wb[c * H + h];
  bias[idx] = s;
}

// out[c][p][q] = in[(p*R+q)*C + c] (swap=0)  or in[(q*R+p)*C + c] (swap=1)
__global__ void k_to_ct(const half_t* __restrict__ in, half_t* __restrict__ out,
                        int R, int C, int swap)
{
  long idx = (long)blockIdx.x * 256 + threadIdx.x;
  long tot = (long)C * R * R;
  if (idx >= tot) return;
  int c = (int)(idx / ((long)R * R));
  long pq = idx % ((long)R * R);
  int p = (int)(pq / R), q = (int)(pq % R);
  long src = swap ? ((long)q * R + p) * C + c : ((long)p * R + q) * C + c;
  out[idx] = in[src];
}

// dst[b*dB + n*dN + c] += src[(b*Nn+n)*C + c]
__global__ void k_add(float* __restrict__ dst, const float* __restrict__ src,
                      long total, long Nn, long C, long dB, long dN)
{
  long idx = (long)blockIdx.x * 256 + threadIdx.x;
  if (idx >= total) return;
  long c = idx % C, t = idx / C;
  long n = t % Nn, b = t / Nn;
  dst[b * dB + n * dN + c] += src[idx];
}

// Weight prep: out (f16, [N][K]) = transpose(in (fp32, [K][N]))
__global__ void k_wt(const float* __restrict__ w, half_t* __restrict__ out,
                     int K, int N)
{
  long idx = (long)blockIdx.x * 256 + threadIdx.x;
  if (idx >= (long)K * N) return;
  long n = idx / K, k = idx % K;
  out[idx] = (half_t)w[k * N + n];
}

// ---------------------------------------------------------------------------
// Host orchestration
// ---------------------------------------------------------------------------
struct AttnP { const float *ln_g,*ln_b,*w_q,*w_k,*w_v,*w_g,*b_g,*w_o,*b_o,*ln_z_g,*ln_z_b,*w_b; };
struct TransP { const float *ln_g,*ln_b,*w1,*b1,*w2,*b2; };
struct OpmP { const float *ln_g,*ln_b,*w_a,*b_a,*w_b,*b_b,*w_o,*b_o; };
struct TriP { const float *ln_g,*ln_b,*w_ag,*b_ag,*w_ap,*b_ap,*w_bg,*b_bg,*w_bp,*b_bp,
                          *w_g,*b_g,*ln2_g,*ln2_b,*w_o,*b_o; };

extern "C" void kernel_launch(void* const* d_in, const int* in_sizes, int n_in,
                              void* d_out, int out_size, void* d_ws, size_t ws_size,
                              hipStream_t stream) {
  const long M_EL = 64L * 256 * 256;     // 4,194,304
  const long Z_EL = 256L * 256 * 128;    // 8,388,608
  const long AR = 8388608;               // arena: 32MB of fp32 / 16M halfs
  if (n_in < 95) return;
  if (ws_size < (size_t)(8 * AR + 524288) * sizeof(float) + (4u << 20)) return;

  int i = 0;
  auto nx = [&](void) { return (const float*)d_in[i++]; };
  const float* in_m = nx();
  const float* in_z = nx();
  AttnP row; row.ln_g=nx(); row.ln_b=nx(); row.w_q=nx(); row.w_k=nx(); row.w_v=nx();
  row.w_g=nx(); row.b_g=nx(); row.w_o=nx(); row.b_o=nx();
  row.ln_z_g=nx(); row.ln_z_b=nx(); row.w_b=nx();
  AttnP col; col.ln_g=nx(); col.ln_b=nx(); col.w_q=nx(); col.w_k=nx(); col.w_v=nx();
  col.w_g=nx(); col.b_g=nx(); col.w_o=nx(); col.b_o=nx();
  TransP mt; mt.ln_g=nx(); mt.ln_b=nx(); mt.w1=nx(); mt.b1=nx(); mt.w2=nx(); mt.b2=nx();
  OpmP opm; opm.ln_g=nx(); opm.ln_b=nx(); opm.w_a=nx(); opm.b_a=nx();
  opm.w_b=nx(); opm.b_b=nx(); opm.w_o=nx(); opm.b_o=nx();
  TriP tmo, tmi;
  TriP* tp[2] = {&tmo, &tmi};
  for (int t = 0; t < 2; ++t) {
    TriP& r = *tp[t];
    r.ln_g=nx(); r.ln_b=nx(); r.w_ag=nx(); r.b_ag=nx(); r.w_ap=nx(); r.b_ap=nx();
    r.w_bg=nx(); r.b_bg=nx(); r.w_bp=nx(); r.b_bp=nx(); r.w_g=nx(); r.b_g=nx();
    r.ln2_g=nx(); r.ln2_b=nx(); r.w_o=nx(); r.b_o=nx();
  }
  AttnP tas, tae;
  AttnP* ap2[2] = {&tas, &tae};
  for (int t = 0; t < 2; ++t) {
    AttnP& r = *ap2[t];
    r.ln_g=nx(); r.ln_b=nx(); r.w_q=nx(); r.w_k=nx(); r.w_v=nx();
    r.w_g=nx(); r.b_g=nx(); r.w_o=nx(); r.b_o=nx(); r.w_b=nx();
  }
  TransP pt; pt.ln_g=nx(); pt.ln_b=nx(); pt.w1=nx(); pt.b1=nx(); pt.w2=nx(); pt.b2=nx();

  float* W = (float*)d_ws;
  float* A0 = W;            float* A1 = W + 1 * AR;  float* A2 = W + 2 * AR;
  float* A3 = W + 3 * AR;   float* A4 = W + 4 * AR;  float* A5 = W + 5 * AR;
  float* A6 = W + 6 * AR;   float* A7 = W + 7 * AR;
  float* BIASB = W + 8 * AR;                          // 524288 fp32
  half_t* A0h = (half_t*)A0; half_t* A1h = (half_t*)A1; half_t* A2h = (half_t*)A2;
  half_t* A3h = (half_t*)A3; half_t* A4h = (half_t*)A4; half_t* A5h = (half_t*)A5;
  half_t* A6h = (half_t*)A6; half_t* A7h = (half_t*)A7;

  // f16 transposed-weight arena
  half_t* wp = (half_t*)(W + 8 * AR + 524288);
  auto wt = [&](const float* w, int K, int N) {
    half_t* d = wp; long tot = (long)K * N; wp += tot;
    k_wt<<<dim3((unsigned)((tot + 255) / 256)), 256, 0, stream>>>(w, d, K, N);
    return d;
  };

  float* mw = (float*)d_out;          // working m, [64,256,256]
  float* zw = mw + M_EL;              // working z, [256,256,128]
  hipMemcpyAsync(mw, in_m, M_EL * sizeof(float), hipMemcpyDeviceToDevice, stream);
  hipMemcpyAsync(zw, in_z, Z_EL * sizeof(float), hipMemcpyDeviceToDevice, stream);

  auto ln = [&](const float* x, const float* g, const float* b, half_t* out,
                long rows, long Nn, long sB, long sN, long sC, int C) {
    k_ln<<<dim3((unsigned)((rows + 7) / 8)), 256, 0, stream>>>(x, g, b, out,
        (int)rows, (int)Nn, sB, sN, sC, C);
  };
  // standard row-major GEMM with pre-transposed f16 weights (Bt ld = K)
  auto gemm = [&](const half_t* A, const half_t* Bt, float* Cf, half_t* Ch,
                  int M, int N, int K, const float* bias, int act, float alpha,
                  const float* mul) {
    k_gemm<<<dim3((N + 63) / 64, (M + 63) / 64, 1), 128, 0, stream>>>(
        A, (long)K, 1L, 0L, Bt, (long)K, 1L, 0L, Cf, Ch, (long)N, 0L,
        bias, mul, M, N, K, alpha, act, 0);
  };
  auto addid = [&](float* dst, const float* src, long rows, int C) {
    long tot = rows * C;
    k_add<<<dim3((unsigned)((tot + 255) / 256)), 256, 0, stream>>>(
        dst, src, tot, 1L, (long)C, (long)C, 0L);
  };
  auto attn = [&](const half_t* q, const half_t* k, const half_t* v,
                  const float* g, const float* bias, half_t* o,
                  int B, int N, int H, int HC) {
    size_t sm = ((size_t)N * 48 + (size_t)32 * (N + 16)) * 2;
    sm = (sm + 15) & ~(size_t)15;
    sm += (size_t)4 * 16 * N * 4;
    k_attn<<<dim3(N / 64, H, B), 128, sm, stream>>>(q, k, v, g, bias, o, N, HC);
  };
  const float QS = 0.17677669529663687f;  // 1/sqrt(32)

  // ---- weight prep (f16, transposed) --------------------------------------
  half_t *row_q=wt(row.w_q,256,256), *row_k=wt(row.w_k,256,256), *row_v=wt(row.w_v,256,256),
         *row_g=wt(row.w_g,256,256), *row_o=wt(row.w_o,256,256);
  half_t *col_q=wt(col.w_q,256,256), *col_k=wt(col.w_k,256,256), *col_v=wt(col.w_v,256,256),
         *col_g=wt(col.w_g,256,256), *col_o=wt(col.w_o,256,256);
  half_t *mt_w1=wt(mt.w1,256,1024), *mt_w2=wt(mt.w2,1024,256);
  half_t *opm_a=wt(opm.w_a,256,32), *opm_b=wt(opm.w_b,256,32), *opm_o=wt(opm.w_o,1024,128);
  half_t *tri_w[2][6];
  for (int t = 0; t < 2; ++t) {
    TriP& r = *tp[t];
    tri_w[t][0]=wt(r.w_ag,128,128); tri_w[t][1]=wt(r.w_ap,128,128);
    tri_w[t][2]=wt(r.w_bg,128,128); tri_w[t][3]=wt(r.w_bp,128,128);
    tri_w[t][4]=wt(r.w_g,128,128);  tri_w[t][5]=wt(r.w_o,128,128);
  }
  half_t *ta_w[2][5];
  for (int t = 0; t < 2; ++t) {
    AttnP& r = *ap2[t];
    ta_w[t][0]=wt(r.w_q,128,128); ta_w[t][1]=wt(r.w_k,128,128); ta_w[t][2]=wt(r.w_v,128,128);
    ta_w[t][3]=wt(r.w_g,128,128); ta_w[t][4]=wt(r.w_o,128,128);
  }
  half_t *pt_w1=wt(pt.w1,128,512), *pt_w2=wt(pt.w2,512,128);

  // ---- 1. MSARowAttentionWithPairBias -------------------------------------
  ln(mw, row.ln_g, row.ln_b, A0h, 16384, 1, 256, 0, 1, 256);
  ln(zw, row.ln_z_g, row.ln_z_b, A1h, 65536, 1, 128, 0, 1, 128);
  k_bias<<<dim3(8 * 65536 / 256), 256, 0, stream>>>(A1h, row.w_b, BIASB, 8, 256, 128);
  gemm(A0h, row_q, nullptr, A2h, 16384, 256, 256, nullptr, 0, QS, nullptr);
  gemm(A0h, row_k, nullptr, A3h, 16384, 256, 256, nullptr, 0, 1.f, nullptr);
  gemm(A0h, row_v, nullptr, A4h, 16384, 256, 256, nullptr, 0, 1.f, nullptr);
  gemm(A0h, row_g, A5, nullptr, 16384, 256, 256, row.b_g, 1, 1.f, nullptr);
  attn(A2h, A3h, A4h, A5, BIASB, A6h, 64, 256, 8, 256);
  gemm(A6h, row_o, A7, nullptr, 16384, 256, 256, row.b_o, 0, 1.f, nullptr);
  addid(mw, A7, 16384, 256);

  // ---- 2. MSAColumnAttention (attention over the 64 sequences) ------------
  ln(mw, col.ln_g, col.ln_b, A0h, 16384, 64, 256, 65536, 1, 256);  // rows = r*64+s
  gemm(A0h, col_q, nullptr, A2h, 16384, 256, 256, nullptr, 0, QS, nullptr);
  gemm(A0h, col_k, nullptr, A3h, 16384, 256, 256, nullptr, 0, 1.f, nullptr);
  gemm(A0h, col_v, nullptr, A4h, 16384, 256, 256, nullptr, 0, 1.f, nullptr);
  gemm(A0h, col_g, A5, nullptr, 16384, 256, 256, col.b_g, 1, 1.f, nullptr);
  attn(A2h, A3h, A4h, A5, nullptr, A6h, 256, 64, 8, 256);
  gemm(A6h, col_o, A7, nullptr, 16384, 256, 256, col.b_o, 0, 1.f, nullptr);
  k_add<<<dim3((16384L * 256 + 255) / 256), 256, 0, stream>>>(
      mw, A7, 16384L * 256, 64L, 256L, 256L, 65536L);  // (r,s)->m[s,r]

  // ---- 3. MSA transition (chunked: hidden 8192x1024) -----------------------
  ln(mw, mt.ln_g, mt.ln_b, A0h, 16384, 1, 256, 0, 1, 256);
  for (long c0 = 0; c0 < 16384; c0 += 8192) {
    gemm(A0h + c0 * 256, mt_w1, nullptr, A1h, 8192, 1024, 256, mt.b1, 2, 1.f, nullptr);
    gemm(A1h, mt_w2, A2, nullptr, 8192, 256, 1024, mt.b2, 0, 1.f, nullptr);
    addid(mw + c0 * 256, A2, 8192, 256);
  }

  // ---- 4. OuterProductMean (chunked over i, 32 rows at a time) -------------
  ln(mw, opm.ln_g, opm.ln_b, A0h, 16384, 1, 256, 0, 1, 256);
  gemm(A0h, opm_a, nullptr, A1h, 16384, 32, 256, opm.b_a, 0, 1.f, nullptr);
  gemm(A0h, opm_b, nullptr, A2h, 16384, 32, 256, opm.b_b, 0, 1.f, nullptr);
  for (int i0 = 0; i0 < 256; i0 += 32) {
    // O[(i,c),(j,d)] = (1/64) sum_s a[s,i,c] b[s,j,d]  -> A3h [1024 x 8192] f16
    k_gemm<<<dim3(8192 / 64, 1024 / 64, 1), 128, 0, stream>>>(
        A1h + (long)i0 * 32, 1L, 8192L, 0L, A2h, 1L, 8192L, 0L,
        nullptr, A3h, 8192L, 0L, nullptr, nullptr, 1024, 8192, 64, 1.f / 64.f, 0, 0);
    // z[(i,j),e] += O.reshape[(i,j),(c,d)] @ w_o + b_o  (opm index mode)
    k_gemm<<<dim3(128 / 64, 8192 / 64, 1), 128, 0, stream>>>(
        A3h, 0L, 0L, 0L, opm_o, 1024L, 1L, 0L,
        A4, nullptr, 128L, 0L, opm.b_o, nullptr, 8192, 128, 1024, 1.f, 0, 1);
    addid(zw + (long)i0 * 256 * 128, A4, 8192, 128);
  }

  // ---- 5/6. Triangle multiplications (outgoing, then incoming) -------------
  for (int t = 0; t < 2; ++t) {
    TriP& tm = *tp[t];
    int incoming = t;  // 0 = 'ikc,jkc', 1 = 'kic,kjc'
    ln(zw, tm.ln_g, tm.ln_b, A0h, 65536, 1, 128, 0, 1, 128);
    gemm(A0h, tri_w[t][0], A1, nullptr, 65536, 128, 128, tm.b_ag, 1, 1.f, nullptr);
    gemm(A0h, tri_w[t][1], nullptr, A2h, 65536, 128, 128, tm.b_ap, 0, 1.f, A1);  // a
    gemm(A0h, tri_w[t][2], A1, nullptr, 65536, 128, 128, tm.b_bg, 1, 1.f, nullptr);
    gemm(A0h, tri_w[t][3], nullptr, A3h, 65536, 128, 128, tm.b_bp, 0, 1.f, A1);  // b
    long tot = 128L * 65536;
    k_to_ct<<<dim3((unsigned)((tot + 255) / 256)), 256, 0, stream>>>(A2h, A4h, 256, 128, incoming);
    k_to_ct<<<dim3((unsigned)((tot + 255) / 256)), 256, 0, stream>>>(A3h, A5h, 256, 128, incoming);
    // X_c = A_c * B_c^T, batched over 128 channels -> fp32 x_t [c][i][j]
    k_gemm<<<dim3(4, 4, 128), 128, 0, stream>>>(
        A4h, 256L, 1L, 65536L, A5h, 256L, 1L, 65536L,
        A6, nullptr, 256L, 65536L, nullptr, nullptr, 256, 256, 256, 1.f, 0, 0);
    ln(A6, tm.ln2_g, tm.ln2_b, A7h, 65536, 1, 1, 0, 65536, 128);    // LN over c
    gemm(A0h, tri_w[t][4], A1, nullptr, 65536, 128, 128, tm.b_g, 1, 1.f, nullptr); // gate
    gemm(A7h, tri_w[t][5], A2, nullptr, 65536, 128, 128, tm.b_o, 0, 1.f, A1);
    addid(zw, A2, 65536, 128);
  }

  // ---- 7/8. Triangle attention (starting node, then ending node on z^T) ----
  for (int t = 0; t < 2; ++t) {
    AttnP& ta = *ap2[t];
    if (t == 0) ln(zw, ta.ln_g, ta.ln_b, A0h, 65536, 1, 128, 0, 1, 128);
    else        ln(zw, ta.ln_g, ta.ln_b, A0h, 65536, 256, 128, 32768, 1, 128); // z^T
    k_bias<<<dim3(4 * 65536 / 256), 256, 0, stream>>>(A0h, ta.w_b, BIASB, 4, 256, 128);
    gemm(A0h, ta_w[t][0], nullptr, A2h, 65536, 128, 128, nullptr, 0, QS, nullptr);
    gemm(A0h, ta_w[t][1], nullptr, A3h, 65536, 128, 128, nullptr, 0, 1.f, nullptr);
    gemm(A0h, ta_w[t][2], nullptr, A4h, 65536, 128, 128, nullptr, 0, 1.f, nullptr);
    gemm(A0h, ta_w[t][3], A5, nullptr, 65536, 128, 128, ta.b_g, 1, 1.f, nullptr);
    attn(A2h, A3h, A4h, A5, BIASB, A6h, 256, 256, 4, 128);
    gemm(A6h, ta_w[t][4], A7, nullptr, 65536, 128, 128, ta.b_o, 0, 1.f, nullptr);
    if (t == 0) addid(zw, A7, 65536, 128);
    else
      k_add<<<dim3((unsigned)((65536L * 128 + 255) / 256)), 256, 0, stream>>>(
          zw, A7, 65536L * 128, 256L, 128L, 128L, 32768L);  // (j,i)->z[i,j]
  }

  // ---- 9. Pair transition (chunked: hidden 16384x512) ----------------------
  ln(zw, pt.ln_g, pt.ln_b, A0h, 65536, 1, 128, 0, 1, 128);
  for (long c0 = 0; c0 < 65536; c0 += 16384) {
    gemm(A0h + c0 * 128, pt_w1, nullptr, A1h, 16384, 512, 128, pt.b1, 2, 1.f, nullptr);
    gemm(A1h, pt_w2, A2, nullptr, 16384, 128, 512, pt.b2, 0, 1.f, nullptr);
    addid(zw + c0 * 128, A2, 16384, 128);
  }
}
